// TestAttentionStaticQuantPatternModel_89223650607626
// MI455X (gfx1250) — matrix operand order repaired and checked by reading, not compiled, not run on hardware
//
#include <hip/hip_runtime.h>
#include <math.h>

// CDNA5 / gfx1250 decode attention (GQA, B=64 S=4096 Hq=32 Hkv=8 D=128) +
// static FP8-e4m3 quant. Memory-bound: streams 2.15 GB fp32 KV once ->
// ~92us @ 23.3 TB/s. One workgroup per (b, kv-head); 2 wave32 waves split S.
// K/V staged global->LDS with global_load_async_to_lds_b128 (ASYNCcnt
// pipeline, K double-buffered, non-temporal), WMMA f32_16x16x32_f16 for
// QK^T and P*V. Only 4 of 16 WMMA rows are real (GQA group); softmax and
// rescale run on real rows only.

typedef __attribute__((ext_vector_type(16))) _Float16 v16h;
typedef __attribute__((ext_vector_type(8)))  _Float16 h8;
typedef __attribute__((ext_vector_type(8)))  float    v8f;

#define B_    64
#define S_    4096
#define HQ_   32
#define HKV_  8
#define D_    128
#define G_    4
#define KVS   (HKV_ * D_)        // 1024 floats between consecutive s
#define WAVES 2
#define SPLIT (S_ / WAVES)       // keys per wave
#define KSTR  132                // K tile row stride (f32) -> bank step 4
#define VSTR  128                // V tile row stride (f32)
#define PT    40                 // P tile row stride (f16), 16B-aligned reloads
#define SM_SCALE 0.08838834764831845f
#define FP8_MAX_ 448.0f

#define ASYNC_WAIT(N) asm volatile("s_wait_asynccnt %0" :: "n"(N) : "memory")

// one wave moves a 32-key x 128-d fp32 chunk global->LDS, 512B per
// instruction, non-temporal (KV is streamed once; don't rinse L2)
__device__ __forceinline__ void async_stage_chunk(const float* __restrict__ g,
                                                  float* __restrict__ ldsbuf,
                                                  int rowStride, int lane) {
  #pragma unroll
  for (int t = 0; t < 32; ++t) {
    unsigned lds = (unsigned)(uintptr_t)(ldsbuf + t * rowStride + lane * 4);
    unsigned long long ga =
        (unsigned long long)(uintptr_t)(g + (size_t)t * KVS + lane * 4);
    asm volatile("global_load_async_to_lds_b128 %0, %1, off th:TH_LOAD_NT"
                 :: "v"(lds), "v"(ga) : "memory");
  }
}

// load 16 consecutive f32 from LDS, convert to f16 vector (B operand K-slice)
__device__ __forceinline__ v16h lds_load_cvt16(const float* p) {
  float4 a = *(const float4*)p;
  float4 b = *(const float4*)(p + 4);
  float4 c = *(const float4*)(p + 8);
  float4 d = *(const float4*)(p + 12);
  v16h r;
  r[0]  = (_Float16)a.x; r[1]  = (_Float16)a.y; r[2]  = (_Float16)a.z; r[3]  = (_Float16)a.w;
  r[4]  = (_Float16)b.x; r[5]  = (_Float16)b.y; r[6]  = (_Float16)b.z; r[7]  = (_Float16)b.w;
  r[8]  = (_Float16)c.x; r[9]  = (_Float16)c.y; r[10] = (_Float16)c.z; r[11] = (_Float16)c.w;
  r[12] = (_Float16)d.x; r[13] = (_Float16)d.y; r[14] = (_Float16)d.z; r[15] = (_Float16)d.w;
  return r;
}

// round-trip f32 -> fp8 e4m3fn -> f32 (RNE, denormal grid 2^-9, clamp +-448)
__device__ __forceinline__ float quant_e4m3(float x) {
  x = fminf(fmaxf(x, -FP8_MAX_), FP8_MAX_);
  if (fabsf(x) < 0.015625f) {                // below min normal 2^-6
    return rintf(x * 512.0f) * (1.0f / 512.0f);
  }
  unsigned u = __float_as_uint(x);
  unsigned lsb = (u >> 20) & 1u;
  u += 0x000FFFFFu + lsb;                    // RNE to 3 mantissa bits
  u &= 0xFFF00000u;
  float y = __uint_as_float(u);
  return fminf(fmaxf(y, -FP8_MAX_), FP8_MAX_);
}

__global__ __launch_bounds__(32 * WAVES, 1)
void attn_decode_fp8_kernel(const float* __restrict__ q,
                            const float* __restrict__ kcache,
                            const float* __restrict__ vcache,
                            const float* __restrict__ qscale,
                            float* __restrict__ out) {
  __shared__ float     sK[WAVES][2][32 * KSTR];  // double-buffered K (f32)
  __shared__ float     sV[WAVES][32 * VSTR];     // V chunk (f32)
  __shared__ _Float16  sP[WAVES][16 * PT];       // P tile (f16)
  __shared__ float     cO[WAVES][G_][D_];        // cross-wave combine
  __shared__ float     cM[WAVES][G_];
  __shared__ float     cL[WAVES][G_];

  const int blk  = blockIdx.x;                   // b*HKV + h
  const int b    = blk / HKV_;
  const int h    = blk % HKV_;
  const int tid  = threadIdx.x;
  const int wv   = tid >> 5;
  const int lane = tid & 31;
  const int lrow = lane & 15;
  const int half = lane >> 4;

  float*    kt0 = sK[wv][0];
  float*    kt1 = sK[wv][1];
  float*    vt  = sV[wv];
  _Float16* pt  = sP[wv];

  // zero the永-garbage P rows {4..7, 12..15} once (A-operand must be finite;
  // these rows only feed O rows that are never read back)
  #pragma unroll
  for (int rr = 0; rr < 8; ++rr) {
    int R = (rr < 4) ? (4 + rr) : (8 + rr);    // 4..7, 12..15
    pt[R * PT + lane] = (_Float16)0.0f;
  }

  // ---- Q in WMMA A-layout, rows >= G zero, pre-scaled by 1/sqrt(D) ----
  // A 16x32 f16, k order chosen here (same as B): lanes 0-15 row=lane, elems 0-15 = K{0..15}+base
  //              lanes 16-31 same row, K{16..31}
  v16h qa[4];
  const float* qrow = q + ((size_t)b * HQ_ + (size_t)(h * G_ + lrow)) * D_;
  #pragma unroll
  for (int kc = 0; kc < 4; ++kc) {
    int dlo = kc * 32 + half * 16;
    #pragma unroll
    for (int j = 0; j < 8; ++j) {
      float v0 = (lrow < G_) ? qrow[dlo + j]      * SM_SCALE : 0.0f;
      float v1 = (lrow < G_) ? qrow[dlo + 8 + j]  * SM_SCALE : 0.0f;
      qa[kc][j]     = (_Float16)v0;
      qa[kc][8 + j] = (_Float16)v1;
    }
  }

  // per-row softmax state for the 4 real slots (row R = r + 8*half)
  const v8f vzero = {0.f, 0.f, 0.f, 0.f, 0.f, 0.f, 0.f, 0.f};
  float m4[G_], l4[G_];
  v8f oacc[8];
  #pragma unroll
  for (int r = 0; r < G_; ++r) { m4[r] = -__builtin_inff(); l4[r] = 0.0f; }
  #pragma unroll
  for (int nt = 0; nt < 8; ++nt) oacc[nt] = vzero;

  const int    sBeg   = wv * SPLIT;
  const int    nIter  = SPLIT / 32;
  const size_t kvbase = ((size_t)b * S_) * KVS + (size_t)h * D_;
  const float* kgbase = kcache + kvbase;
  const float* vgbase = vcache + kvbase;

  // prologue: launch K(0) then V(0); 64 async b128 in flight
  async_stage_chunk(kgbase + (size_t)sBeg * KVS, kt0, KSTR, lane);
  async_stage_chunk(vgbase + (size_t)sBeg * KVS, vt,  VSTR, lane);

  for (int i = 0; i < nIter; ++i) {
    const int s0 = sBeg + i * 32;
    float* ktCur = (i & 1) ? kt1 : kt0;
    float* ktNxt = (i & 1) ? kt0 : kt1;
    // last iteration: dummy re-stage of first chunk keeps wait math uniform
    const int sNxt = (i + 1 < nIter) ? (s0 + 32) : sBeg;

    ASYNC_WAIT(32);     // oldest 32 = K(i) landed (async loads retire in order)
    async_stage_chunk(kgbase + (size_t)sNxt * KVS, ktNxt, KSTR, lane);

    // ---- scores = Q K^T : two 16x16 C tiles (keys 0-15, 16-31) ----
    // B 32x16 f16, k order chosen here (as in Q): lanes 0-15 col=lane hold K{0..15}, lanes 16-31 hold K{16..31}
    v8f acc0 = vzero, acc1 = vzero;
    #pragma unroll
    for (int kc = 0; kc < 4; ++kc) {
      int d0 = kc * 32 + half * 16;
      v16h b0 = lds_load_cvt16(&ktCur[lrow * KSTR + d0]);
      v16h b1 = lds_load_cvt16(&ktCur[(16 + lrow) * KSTR + d0]);
      acc0 = __builtin_amdgcn_wmma_f32_16x16x32_f16(false, qa[kc], false, b0,
                                                    (short)0, acc0, false, false);
      acc1 = __builtin_amdgcn_wmma_f32_16x16x32_f16(false, qa[kc], false, b1,
                                                    (short)0, acc1, false, false);
    }

    // ---- online softmax, real row slots only (r=0..3) ----
    // C layout: VGPR slot r = row r (lanes 0-15) / row r+8 (lanes 16-31)
    float scl[G_];
    #pragma unroll
    for (int r = 0; r < G_; ++r) {
      float cmax = fmaxf(acc0[r], acc1[r]);
      #pragma unroll
      for (int off = 8; off >= 1; off >>= 1)
        cmax = fmaxf(cmax, __shfl_xor(cmax, off, 32));   // 16-lane group max
      float mn = fmaxf(m4[r], cmax);
      scl[r] = __expf(m4[r] - mn);
      m4[r] = mn;
      float p0 = __expf(acc0[r] - mn);
      float p1 = __expf(acc1[r] - mn);
      float rs = p0 + p1;
      #pragma unroll
      for (int off = 8; off >= 1; off >>= 1)
        rs += __shfl_xor(rs, off, 32);                   // 16-lane group sum
      l4[r] = l4[r] * scl[r] + rs;
      int R = r + 8 * half;                              // rows 0-3 / 8-11
      pt[R * PT + lrow]      = (_Float16)p0;             // key = lane&15
      pt[R * PT + 16 + lrow] = (_Float16)p1;             // key = 16+(lane&15)
    }
    __builtin_amdgcn_wave_barrier();                     // DS in-order per wave

    // ---- P back in A-layout (LDS transpose) ----
    v16h pa;
    {
      h8 plo = *(const h8*)&pt[lrow * PT + half * 16];
      h8 phi = *(const h8*)&pt[lrow * PT + half * 16 + 8];
      #pragma unroll
      for (int j = 0; j < 8; ++j) { pa[j] = plo[j]; pa[8 + j] = phi[j]; }
    }

    ASYNC_WAIT(32);     // oldest 32 = V(i) landed (K(i+1) still in flight)

    // ---- O = O*rescale + P V : 8 N-tiles over D ----
    // V B-operand from LDS: col = d (lane-consecutive -> conflict-free),
    // K = keys half*16 .. half*16+15
    #pragma unroll
    for (int nt = 0; nt < 8; ++nt) {
      int dcol = nt * 16 + lrow;
      v16h vB;
      #pragma unroll
      for (int j = 0; j < 16; ++j)
        vB[j] = (_Float16)vt[(half * 16 + j) * VSTR + dcol];
      v8f c = oacc[nt];
      #pragma unroll
      for (int r = 0; r < G_; ++r) c[r] *= scl[r];       // real rows only
      oacc[nt] = __builtin_amdgcn_wmma_f32_16x16x32_f16(false, pa, false, vB,
                                                        (short)0, c, false, false);
    }

    // refill V for the next chunk (safe: all V reads retired by wmma waits)
    async_stage_chunk(vgbase + (size_t)sNxt * KVS, vt, VSTR, lane);
  }

  // ---- cross-wave flash combine (only rows 0..G-1 are real) ----
  if (half == 0) {
    #pragma unroll
    for (int r = 0; r < G_; ++r) {
      #pragma unroll
      for (int nt = 0; nt < 8; ++nt)
        cO[wv][r][nt * 16 + lrow] = oacc[nt][r];
    }
    if (lane == 0) {
      #pragma unroll
      for (int r = 0; r < G_; ++r) { cM[wv][r] = m4[r]; cL[wv][r] = l4[r]; }
    }
  }
  __syncthreads();

  const float inv_s = 1.0f / qscale[0];
  #pragma unroll
  for (int r = 0; r < G_; ++r) {
    float mstar = fmaxf(cM[0][r], cM[1][r]);
    float e0 = __expf(cM[0][r] - mstar);
    float e1 = __expf(cM[1][r] - mstar);
    float den = e0 * cL[0][r] + e1 * cL[1][r];
    #pragma unroll
    for (int dd = 0; dd < D_ / (32 * WAVES); ++dd) {
      int d = tid + dd * 32 * WAVES;
      float num = e0 * cO[0][r][d] + e1 * cO[1][r][d];
      float o = num / den;
      out[(size_t)b * (HQ_ * D_) + (size_t)(h * G_ + r) * D_ + d] =
          quant_e4m3(o * inv_s);
    }
  }
}

extern "C" void kernel_launch(void* const* d_in, const int* in_sizes, int n_in,
                              void* d_out, int out_size, void* d_ws, size_t ws_size,
                              hipStream_t stream) {
  (void)in_sizes; (void)n_in; (void)out_size; (void)d_ws; (void)ws_size;
  const float* q  = (const float*)d_in[0];
  const float* k  = (const float*)d_in[1];
  const float* v  = (const float*)d_in[2];
  const float* qs = (const float*)d_in[3];
  float* out = (float*)d_out;
  dim3 grid(B_ * HKV_);
  dim3 block(32 * WAVES);
  hipLaunchKernelGGL(attn_decode_fp8_kernel, grid, block, 0, stream,
                     q, k, v, qs, out);
}
